// LSTMModel_with_theta_32014686224942
// MI455X (gfx1250) — 8-bit matrix operand order repaired and checked by reading, compiled, not run on hardware
//
#include <hip/hip_runtime.h>
#include <hip/hip_bf16.h>
#include <stdint.h>


typedef int      v4i   __attribute__((ext_vector_type(4)));
typedef int      v8i   __attribute__((ext_vector_type(8)));
typedef float    v8f   __attribute__((ext_vector_type(8)));
typedef unsigned u32x4 __attribute__((ext_vector_type(4)));
typedef int      i32x4 __attribute__((ext_vector_type(4)));
typedef int      i32x8 __attribute__((ext_vector_type(8)));

#define B_    64
#define T_    2048
#define H_    256
#define G4H   1024
#define TC    128
#define NCHUNK (T_ / TC)

#define OFF_WH0   (0u)
#define OFF_WH1   (OFF_WH0 + 262144u)
#define OFF_WX1   (OFF_WH1 + 262144u)
#define OFF_C0    (OFF_WX1 + 262144u)
#define OFF_H0S   (OFF_C0  + 65536u)
#define OFF_C1    (OFF_H0S + 16384u)
#define OFF_H1S   (OFF_C1  + 65536u)
#define OFF_HSUM  (OFF_H1S + 16384u)
#define OFF_H0C   (OFF_HSUM + 65536u)
#define OFF_XG1   (OFF_H0C + (unsigned)(TC*4*4096))
#define OFF_Z     (OFF_XG1 + (unsigned)(TC*64*1024*4))
#define OFF_A1    (OFF_Z   + 131072u)
#define OFF_A2    (OFF_A1  + 65536u)

#define SMEM_BYTES (262144 + 32768 + 4096)

#define AS_GLOBAL __attribute__((address_space(1)))
#define AS_LDS    __attribute__((address_space(3)))

#if __has_builtin(__builtin_amdgcn_global_load_async_to_lds_b128)
#define HAVE_ASYNC_LDS 1
__device__ __forceinline__ void async_g2l_b128(const void* g, void* l) {
  __builtin_amdgcn_global_load_async_to_lds_b128(
      (AS_GLOBAL v4i*)(uintptr_t)g,
      (AS_LDS v4i*)(unsigned)(uintptr_t)l, 0, 0);
}
#endif
#if __has_builtin(__builtin_amdgcn_tensor_load_to_lds)
#define HAVE_TDM 1
#endif

__device__ __forceinline__ void wait_async0() {
#if __has_builtin(__builtin_amdgcn_s_wait_asynccnt)
  __builtin_amdgcn_s_wait_asynccnt(0);
#else
  asm volatile("s_wait_asynccnt 0x0" ::: "memory");
#endif
}
__device__ __forceinline__ void wait_tensor0() {
#if __has_builtin(__builtin_amdgcn_s_wait_tensorcnt)
  __builtin_amdgcn_s_wait_tensorcnt(0);
#else
  asm volatile("s_wait_tensorcnt 0x0" ::: "memory");
#endif
}

__device__ __forceinline__ void tdm_load_wh(const void* gsrc, void* lds) {
#if defined(HAVE_TDM)
  unsigned ldsoff = (unsigned)(uintptr_t)lds;
  uint64_t ga = (uint64_t)(uintptr_t)gsrc;
  u32x4 g0 = {1u, ldsoff, (unsigned)ga, (unsigned)(ga >> 32) | (2u << 30)};
  i32x8 g1 = {(int)(3u << 16), (int)(128u << 16), (int)(256u << 16),
              (int)(128u << 16), 256, 128, 0, 0};
  i32x4 gz = {0, 0, 0, 0};
#if __clang_major__ >= 23
  i32x8 gz8 = {0, 0, 0, 0, 0, 0, 0, 0};
  __builtin_amdgcn_tensor_load_to_lds(g0, g1, gz, gz, gz8, 0);
#else
  __builtin_amdgcn_tensor_load_to_lds(g0, g1, gz, gz, 0);
#endif
#else
  (void)gsrc; (void)lds;
#endif
}

__device__ __forceinline__ unsigned pk4_fp8(float a, float b, float c, float d) {
  int lo = __builtin_amdgcn_cvt_pk_fp8_f32(a, b, 0, false);
  int v  = __builtin_amdgcn_cvt_pk_fp8_f32(c, d, lo, true);
  return (unsigned)v;
}

__device__ __forceinline__ float sigm(float v) {
  return __builtin_amdgcn_rcpf(1.0f + __expf(-v));
}
__device__ __forceinline__ float tanh_(float v) {
#if __has_builtin(__builtin_amdgcn_tanhf)
  return __builtin_amdgcn_tanhf(v);
#elif __has_builtin(__builtin_amdgcn_tanh_f32)
  return __builtin_amdgcn_tanh_f32(v);
#else
  return 1.0f - 2.0f * __builtin_amdgcn_rcpf(1.0f + __expf(2.0f * v));
#endif
}

__global__ void k_prep(const float* __restrict__ Wh0,
                       const float* __restrict__ Wx1,
                       const float* __restrict__ Wh1,
                       unsigned char* __restrict__ ws) {
  int idx = blockIdx.x * blockDim.x + threadIdx.x;
  int mat = idx >> 16;
  int r   = idx & 0xFFFF;
  int n   = r >> 6;
  int k   = (r & 63) * 4;
  const float* W = (mat == 0) ? Wh0 : ((mat == 1) ? Wx1 : Wh1);
  unsigned base  = (mat == 0) ? OFF_WH0 : ((mat == 1) ? OFF_WX1 : OFF_WH1);
  unsigned pk = pk4_fp8(W[(k + 0) * G4H + n], W[(k + 1) * G4H + n],
                        W[(k + 2) * G4H + n], W[(k + 3) * G4H + n]);
  int nt = n >> 4, kb = k >> 6, krel = k & 63;
  int hi = (krel >> 3) & 1;
  int v  = ((krel >> 4) << 1) | ((krel >> 2) & 1);
  int lane = hi * 16 + (n & 15);
  *(unsigned*)(ws + base + (unsigned)((nt * 4 + kb) * 1024 + lane * 32 + v * 4)) = pk;
}

__global__ void k_init(unsigned char* __restrict__ ws) {
  int id = blockIdx.x * blockDim.x + threadIdx.x;
  ((unsigned*)(ws + OFF_C0))[id] = 0u;
}

template <int LAYER>
__global__ __launch_bounds__(256, 1) void k_recur(const float* __restrict__ x,
                                                  const float* __restrict__ Wx0,
                                                  const float* __restrict__ b0,
                                                  unsigned char* __restrict__ ws,
                                                  int chunk) {
  extern __shared__ unsigned char smem[];
  unsigned char* whs    = smem;
  float*         gatesL = (float*)(smem + 262144);
  unsigned char* hb     = smem + 262144 + 32768;

  const int tid  = threadIdx.x;
  const int wg   = blockIdx.x;
  const int wv   = tid >> 5;
  const int lane = tid & 31;
  const int hi   = lane >> 4;
  const int ln16 = lane & 15;
  const int m    = tid >> 4;
  const int c0   = (tid & 15) * 16;
  const int rowg = wg * 16 + m;

  const size_t wbase = (LAYER == 0) ? (size_t)OFF_WH0 : (size_t)OFF_WH1;
#if defined(HAVE_TDM)
  if (tid == 0) tdm_load_wh(ws + wbase, whs);
#elif defined(HAVE_ASYNC_LDS)
  for (int i = 0; i < 64; ++i) {
    int idx = i * 256 + tid;
    async_g2l_b128(ws + wbase + (size_t)idx * 16, whs + idx * 16);
  }
#else
  #pragma unroll 4
  for (int i = 0; i < 64; ++i) {
    int idx = i * 256 + tid;
    ((uint4*)whs)[idx] = ((const uint4*)(ws + wbase))[idx];
  }
#endif
  const size_t hsoff = ((LAYER == 0) ? (size_t)OFF_H0S : (size_t)OFF_H1S) + (size_t)wg * 4096;
#if defined(HAVE_ASYNC_LDS)
  async_g2l_b128(ws + hsoff + (size_t)tid * 16, hb + tid * 16);
  wait_async0();
#else
  *(uint4*)(hb + tid * 16) = *(const uint4*)(ws + hsoff + tid * 16);
#endif
  wait_tensor0();

  float* cst = (float*)(ws + ((LAYER == 0) ? (size_t)OFF_C0 : (size_t)OFF_C1)) +
               (size_t)(wg * 256 + tid) * 16;
  float c[16];
  #pragma unroll
  for (int u = 0; u < 16; ++u) c[u] = cst[u];

  float wxi[16], wxf[16], wxg[16], wxo[16];
  float bbi[16], bbf[16], bbg[16], bbo[16];
  if (LAYER == 0) {
    #pragma unroll
    for (int u = 0; u < 16; ++u) {
      wxi[u] = Wx0[c0 + u];         bbi[u] = b0[c0 + u];
      wxf[u] = Wx0[256 + c0 + u];   bbf[u] = b0[256 + c0 + u];
      wxg[u] = Wx0[512 + c0 + u];   bbg[u] = b0[512 + c0 + u];
      wxo[u] = Wx0[768 + c0 + u];   bbo[u] = b0[768 + c0 + u];
    }
  }
  float hs[16];
  #pragma unroll
  for (int u = 0; u < 16; ++u) hs[u] = 0.0f;
  float sI[16], aC[16];
  const float* xg1base = (const float*)(ws + OFF_XG1);

  for (int ts = 0; ts < TC; ++ts) {
    const int t = chunk * TC + ts;
    __syncthreads();
    float xv = 0.0f;
    if (LAYER == 0) xv = x[(size_t)rowg * T_ + t];
    const float* xg = xg1base + ((size_t)(ts * 4 + wg) * 16 + m) * 1024;

    #pragma unroll
    for (int half = 0; half < 2; ++half) {
      v8f acc[4];
      #pragma unroll
      for (int i = 0; i < 4; ++i)
        #pragma unroll
        for (int r = 0; r < 8; ++r) acc[i][r] = 0.0f;

      #pragma unroll
      for (int kb = 0; kb < 4; ++kb) {
        v8i af;
        {
          const int ab = ln16 * 256 + kb * 64 + hi * 8;
          uint2 p0 = *(const uint2*)(hb + ab);
          uint2 p1 = *(const uint2*)(hb + ab + 16);
          uint2 p2 = *(const uint2*)(hb + ab + 32);
          uint2 p3 = *(const uint2*)(hb + ab + 48);
          af.s0 = p0.x; af.s1 = p0.y; af.s2 = p1.x; af.s3 = p1.y;
          af.s4 = p2.x; af.s5 = p2.y; af.s6 = p3.x; af.s7 = p3.y;
        }
        #pragma unroll
        for (int i = 0; i < 4; ++i) {
          const int nt = half * 32 + wv * 4 + i;
          const int bbad = (nt * 4 + kb) * 1024 + lane * 32;
          uint4 q0 = *(const uint4*)(whs + bbad);
          uint4 q1 = *(const uint4*)(whs + bbad + 16);
          v8i bf;
          bf.s0 = q0.x; bf.s1 = q0.y; bf.s2 = q0.z; bf.s3 = q0.w;
          bf.s4 = q1.x; bf.s5 = q1.y; bf.s6 = q1.z; bf.s7 = q1.w;
          acc[i] = __builtin_amdgcn_wmma_f32_16x16x64_fp8_fp8(
                       af, bf, (short)0, acc[i], false, false);
        }
      }
      if (half == 1) __syncthreads();
      #pragma unroll
      for (int i = 0; i < 4; ++i) {
        const int colL = (wv * 4 + i) * 16 + ln16;
        #pragma unroll
        for (int r = 0; r < 8; ++r)
          gatesL[(r + 8 * hi) * 512 + colL] = acc[i][r];
      }
      __syncthreads();

      if (half == 0) {
        #pragma unroll
        for (int u = 0; u < 16; ++u) {
          float iv = gatesL[m * 512 + c0 + u];
          float fv = gatesL[m * 512 + 256 + c0 + u];
          if (LAYER == 0) { iv += xv * wxi[u] + bbi[u]; fv += xv * wxf[u] + bbf[u]; }
          else            { iv += xg[c0 + u];           fv += xg[256 + c0 + u]; }
          sI[u] = sigm(iv);
          aC[u] = sigm(fv) * c[u];
        }
      } else {
        float hv[16];
        #pragma unroll
        for (int u = 0; u < 16; ++u) {
          float gv = gatesL[m * 512 + c0 + u];
          float ov = gatesL[m * 512 + 256 + c0 + u];
          if (LAYER == 0) { gv += xv * wxg[u] + bbg[u]; ov += xv * wxo[u] + bbo[u]; }
          else            { gv += xg[512 + c0 + u];     ov += xg[768 + c0 + u]; }
          float cn = aC[u] + sI[u] * tanh_(gv);
          c[u] = cn;
          float h1 = sigm(ov) * tanh_(cn);
          hv[u] = h1;
          if (LAYER == 1) hs[u] += h1;
        }
        uint4 hq;
        hq.x = pk4_fp8(hv[0],  hv[1],  hv[2],  hv[3]);
        hq.y = pk4_fp8(hv[4],  hv[5],  hv[6],  hv[7]);
        hq.z = pk4_fp8(hv[8],  hv[9],  hv[10], hv[11]);
        hq.w = pk4_fp8(hv[12], hv[13], hv[14], hv[15]);
        *(uint4*)(hb + m * 256 + c0) = hq;
        if (LAYER == 0)
          *(uint4*)(ws + OFF_H0C + (size_t)(ts * 4 + wg) * 4096 + m * 256 + c0) = hq;
      }
    }
  }

  __syncthreads();
  #pragma unroll
  for (int u = 0; u < 16; ++u) cst[u] = c[u];
  *(uint4*)(ws + hsoff + tid * 16) = *(const uint4*)(hb + tid * 16);
  if (LAYER == 1) {
    float* hp = (float*)(ws + OFF_HSUM) + (size_t)rowg * 256 + c0;
    #pragma unroll
    for (int u = 0; u < 16; ++u) hp[u] += hs[u];
  }
}

__global__ __launch_bounds__(256) void k_xg1(const float* __restrict__ b1,
                                             unsigned char* __restrict__ ws) {
  const int tid = threadIdx.x;
  const int lane = tid & 31, hi = lane >> 4, ln16 = lane & 15;
  const int gw = blockIdx.x * 8 + (tid >> 5);
  const int nt = gw & 63;
  const int mt = gw >> 6;
  const unsigned char* A  = ws + OFF_H0C + (size_t)mt * 4096;
  const unsigned char* Bw = ws + OFF_WX1;
  v8f acc;
  #pragma unroll
  for (int r = 0; r < 8; ++r) acc[r] = 0.0f;
  #pragma unroll
  for (int kb = 0; kb < 4; ++kb) {
    const int ab = ln16 * 256 + kb * 64 + hi * 8;
    uint2 p0 = *(const uint2*)(A + ab);
    uint2 p1 = *(const uint2*)(A + ab + 16);
    uint2 p2 = *(const uint2*)(A + ab + 32);
    uint2 p3 = *(const uint2*)(A + ab + 48);
    v8i af;
    af.s0 = p0.x; af.s1 = p0.y; af.s2 = p1.x; af.s3 = p1.y;
    af.s4 = p2.x; af.s5 = p2.y; af.s6 = p3.x; af.s7 = p3.y;
    const int bbad = (nt * 4 + kb) * 1024 + lane * 32;
    uint4 q0 = *(const uint4*)(Bw + bbad);
    uint4 q1 = *(const uint4*)(Bw + bbad + 16);
    v8i bf;
    bf.s0 = q0.x; bf.s1 = q0.y; bf.s2 = q0.z; bf.s3 = q0.w;
    bf.s4 = q1.x; bf.s5 = q1.y; bf.s6 = q1.z; bf.s7 = q1.w;
    acc = __builtin_amdgcn_wmma_f32_16x16x64_fp8_fp8(af, bf, (short)0, acc, false, false);
  }
  const int col = nt * 16 + ln16;
  const float bias = b1[col];
  float* outp = (float*)(ws + OFF_XG1) + (size_t)mt * 16 * 1024;
  #pragma unroll
  for (int r = 0; r < 8; ++r)
    outp[(size_t)(r + 8 * hi) * 1024 + col] = acc[r] + bias;
}

__global__ void k_mlp_a(const float* __restrict__ theta, const float* __restrict__ Wtp,
                        const float* __restrict__ btp, unsigned char* __restrict__ ws) {
  int id = blockIdx.x * blockDim.x + threadIdx.x;
  int b = id >> 9, col = id & 511;
  float v;
  if (col < 256) {
    v = ((const float*)(ws + OFF_HSUM))[b * 256 + col] * (1.0f / (float)T_);
  } else {
    int k = col - 256;
    v = btp[k];
    #pragma unroll
    for (int d = 0; d < 5; ++d) v += theta[b * 5 + d] * Wtp[d * 256 + k];
  }
  ((float*)(ws + OFF_Z))[id] = v;
}
__global__ void k_mlp_b(const float* __restrict__ Wl1, const float* __restrict__ bl1,
                        unsigned char* __restrict__ ws) {
  int id = blockIdx.x * blockDim.x + threadIdx.x;
  int b = id >> 8, n = id & 255;
  const float* z = (const float*)(ws + OFF_Z) + b * 512;
  float s = bl1[n];
  for (int k = 0; k < 512; ++k) s += z[k] * Wl1[k * 256 + n];
  s = s > 0.0f ? s : (__expf(s) - 1.0f);
  ((float*)(ws + OFF_A1))[id] = s;
}
__global__ void k_mlp_c(const float* __restrict__ Wl2, const float* __restrict__ bl2,
                        unsigned char* __restrict__ ws) {
  int id = blockIdx.x * blockDim.x + threadIdx.x;
  int b = id >> 8, n = id & 255;
  const float* a1 = (const float*)(ws + OFF_A1) + b * 256;
  float s = bl2[n];
  for (int k = 0; k < 256; ++k) s += a1[k] * Wl2[k * 256 + n];
  s = s > 0.0f ? s : (__expf(s) - 1.0f);
  ((float*)(ws + OFF_A2))[id] = s;
}
__global__ void k_mlp_d(const float* __restrict__ Wo, const float* __restrict__ bo,
                        unsigned char* __restrict__ ws, float* __restrict__ out) {
  int b = threadIdx.x;
  const float* a2 = (const float*)(ws + OFF_A2) + b * 256;
  float s = bo[0];
  for (int k = 0; k < 256; ++k) s += a2[k] * Wo[k];
  out[b] = s;
}

extern "C" void kernel_launch(void* const* d_in, const int* in_sizes, int n_in,
                              void* d_out, int out_size, void* d_ws, size_t ws_size,
                              hipStream_t stream) {
  (void)in_sizes; (void)n_in; (void)out_size; (void)ws_size;
  const float* x   = (const float*)d_in[0];
  const float* th  = (const float*)d_in[1];
  const float* Wx0 = (const float*)d_in[2];
  const float* Wh0 = (const float*)d_in[3];
  const float* b0  = (const float*)d_in[4];
  const float* Wx1 = (const float*)d_in[5];
  const float* Wh1 = (const float*)d_in[6];
  const float* b1  = (const float*)d_in[7];
  const float* Wtp = (const float*)d_in[8];
  const float* btp = (const float*)d_in[9];
  const float* Wl1 = (const float*)d_in[10];
  const float* bl1 = (const float*)d_in[11];
  const float* Wl2 = (const float*)d_in[12];
  const float* bl2 = (const float*)d_in[13];
  const float* Wo  = (const float*)d_in[14];
  const float* bo  = (const float*)d_in[15];
  unsigned char* ws = (unsigned char*)d_ws;
  float* out = (float*)d_out;

  (void)hipFuncSetAttribute((const void*)k_recur<0>,
                            hipFuncAttributeMaxDynamicSharedMemorySize, SMEM_BYTES);
  (void)hipFuncSetAttribute((const void*)k_recur<1>,
                            hipFuncAttributeMaxDynamicSharedMemorySize, SMEM_BYTES);

  k_prep<<<768, 256, 0, stream>>>(Wh0, Wx1, Wh1, ws);
  k_init<<<224, 256, 0, stream>>>(ws);
  for (int c = 0; c < NCHUNK; ++c) {
    k_recur<0><<<4, 256, SMEM_BYTES, stream>>>(x, Wx0, b0, ws, c);
    k_xg1<<<TC * 32, 256, 0, stream>>>(b1, ws);
    k_recur<1><<<4, 256, SMEM_BYTES, stream>>>(x, Wx0, b0, ws, c);
  }
  k_mlp_a<<<128, 256, 0, stream>>>(th, Wtp, btp, ws);
  k_mlp_b<<<64, 256, 0, stream>>>(Wl1, bl1, ws);
  k_mlp_c<<<64, 256, 0, stream>>>(Wl2, bl2, ws);
  k_mlp_d<<<1, 64, 0, stream>>>(Wo, bo, ws, out);
}
